// TransformerDecoderLayer_67619965108677
// MI455X (gfx1250) — matrix operand order repaired and checked by reading, not compiled, not run on hardware
//
#include <hip/hip_runtime.h>

// MI455X / gfx1250, wave32. WMMA f32_16x16x32_f16 everywhere.
// Relation branch (103 GFLOP, 268MB streamed once, ~380 FLOP/byte -> compute
// bound): fused kernel with TDM (tensor_load_to_lds) double-buffered staging of
// the relative_pos stream, 2x2-register-blocked WMMA GEMM chain, per-feature
// softmax-sums via ds_add_f32 LDS atomics.

typedef _Float16 h16;
typedef __attribute__((ext_vector_type(16))) _Float16 v16h;
typedef __attribute__((ext_vector_type(8)))  _Float16 v8h;
typedef __attribute__((ext_vector_type(8)))  float    v8f;
typedef __attribute__((ext_vector_type(4)))  unsigned int v4u;
typedef __attribute__((ext_vector_type(8)))  int v8i;
typedef __attribute__((ext_vector_type(4)))  int v4i;

#define DD 256
#define NH 8
#define HDim 32
#define FFD 1024
#define QQ 128
#define CC 512
#define BB 4

#if defined(__has_builtin)
#if __has_builtin(__builtin_amdgcn_tensor_load_to_lds)
#define USE_TDM 1
#endif
#endif
#ifndef USE_TDM
#define USE_TDM 0
#endif

__device__ __forceinline__ v8f wmma_f16(v16h a, v16h b, v8f c) {
  return __builtin_amdgcn_wmma_f32_16x16x32_f16(false, a, false, b, (short)0, c,
                                                false, false);
}

// A fragment (16x32, f16). lane L holds row M=L%16; halves j = K = kb + off + j,
// off = (L<16 ? 0 : 16) (contiguous). Row-major source.
__device__ __forceinline__ v16h load_afrag(const h16* __restrict__ A, int rs, int kb) {
  int l = threadIdx.x & 31;
  const h16* p = A + (l & 15) * rs + kb + ((l < 16) ? 0 : 16);
  union { v16h v; v8h h[2]; } u;
  u.h[0] = *(const v8h*)p;
  u.h[1] = *(const v8h*)(p + 8);
  return u.v;
}

// B fragment (32x16, f16) for out = A @ W^T with W row-major (N x K):
// lane L holds column N=L%16; halves j = K = kb + (L<16?0:16) + j (contiguous).
__device__ __forceinline__ v16h load_bfrag(const h16* __restrict__ W, int ns, int kb) {
  int l = threadIdx.x & 31;
  const h16* p = W + (l & 15) * ns + kb + ((l < 16) ? 0 : 16);
  union { v16h v; v8h h[2]; } u;
  u.h[0] = *(const v8h*)p;
  u.h[1] = *(const v8h*)(p + 8);
  return u.v;
}

// Strided B fragment: element (n,k) at W[n*ns + k*ks] (used for P@V).
__device__ __forceinline__ v16h load_bfrag_s(const h16* __restrict__ W, int ns,
                                             int ks, int kb) {
  int l = threadIdx.x & 31;
  const h16* p = W + (long)(l & 15) * ns;
  int k0 = kb + ((l < 16) ? 0 : 16);
  v16h r;
#pragma unroll
  for (int j = 0; j < 16; ++j) r[j] = p[(long)(k0 + j) * ks];
  return r;
}

#if USE_TDM
// 2D tile DMA: rows x rowlen f32 elements, row stride rowstride (elems),
// global -> LDS (contiguous). D# layout per CDNA5 ISA ch.8 (group0/group1).
__device__ __forceinline__ void tdm_load_2d(const float* gsrc, unsigned lds_off,
                                            int rows, int rowlen, int rowstride) {
  unsigned long long ga = (unsigned long long)(uintptr_t)gsrc;
  v4u g0;
  g0[0] = 1u;                                   // count=1 (valid), user mode
  g0[1] = lds_off;                              // D#.lds_addr (bytes)
  g0[2] = (unsigned)ga;                         // global_addr[31:0]
  g0[3] = (unsigned)((ga >> 32) & 0x1FFFFFFu) | (2u << 30);  // addr[56:32]|type=2
  v8i g1;
  g1[0] = (int)(2u << 16);                      // data_size = 4B, no multicast
  g1[1] = (int)((unsigned)rowlen << 16);        // tensor_dim0[15:0] @ [31:16]
  g1[2] = (int)(((unsigned)rowlen >> 16) | (0xFFFFu << 16)); // dim0 hi | dim1 lo
  g1[3] = (int)((unsigned)rowlen << 16);        // dim1 hi=0 | tile_dim0 @ [31:16]
  g1[4] = rows;                                 // tile_dim1 (tile_dim2 = 0)
  g1[5] = rowstride;                            // tensor_dim0_stride[31:0]
  g1[6] = 0;
  g1[7] = 0;
  v4i z = {0, 0, 0, 0};
#if __has_include(<hip/amd_detail/amd_gfx1250_TDM.h>)
  v8i z8 = {0, 0, 0, 0, 0, 0, 0, 0};
  __builtin_amdgcn_tensor_load_to_lds(g0, g1, z, z, z8, 0);
#else
  __builtin_amdgcn_tensor_load_to_lds(g0, g1, z, z, 0);
#endif
}
#endif

// ---------------- generic WMMA GEMM: out = act( scale*(A@W^T) + bias + res ) --
__global__ void __launch_bounds__(256) k_gemm(
    const h16* __restrict__ A, long a_off, long a_bs, long a_hs, int a_rs,
    const h16* __restrict__ W, long w_off, long w_bs, long w_hs, int w_ns, int w_ks,
    const float* __restrict__ bias,
    const float* __restrict__ res, long r_off, long r_bs, long r_hs, int r_rs,
    float* __restrict__ outF, h16* __restrict__ outH,
    long o_off, long o_bs, long o_hs, int o_rs,
    int M, int N, int K, int Hn, int relu, float scale) {
  int z = blockIdx.z;
  int b = z / Hn, h = z % Hn;
  int wid = threadIdx.x >> 5, l = threadIdx.x & 31;
  int mt = blockIdx.y * 8 + wid;
  int nt = blockIdx.x;
  if (mt * 16 >= M) return;
  const h16* Ab = A + a_off + b * a_bs + h * a_hs + (long)mt * 16 * a_rs;
  const h16* Wb = W + w_off + b * w_bs + h * w_hs + (long)nt * 16 * w_ns;
  v8f acc = {};
  if (w_ks == 1) {
    for (int kb = 0; kb < K; kb += 32)
      acc = wmma_f16(load_afrag(Ab, a_rs, kb), load_bfrag(Wb, w_ns, kb), acc);
  } else {
    for (int kb = 0; kb < K; kb += 32)
      acc = wmma_f16(load_afrag(Ab, a_rs, kb), load_bfrag_s(Wb, w_ns, w_ks, kb), acc);
  }
  int n = nt * 16 + (l & 15);
  int m0 = mt * 16 + ((l < 16) ? 0 : 8);
  float bv = bias ? bias[n] : 0.f;
#pragma unroll
  for (int j = 0; j < 8; ++j) {
    int m = m0 + j;
    float v = acc[j] * scale + bv;
    if (res) v += res[r_off + b * r_bs + h * r_hs + (long)m * r_rs + n];
    if (relu) v = v > 0.f ? v : 0.f;
    long oi = o_off + b * o_bs + h * o_hs + (long)m * o_rs + n;
    if (outF) outF[oi] = v;
    if (outH) outH[oi] = (h16)v;
  }
}

// ---------------- layernorm over rows of 256 ----------------
__global__ void __launch_bounds__(256) k_ln(const float* __restrict__ x,
                                            const float* __restrict__ g,
                                            const float* __restrict__ be,
                                            float* __restrict__ y) {
  __shared__ float red[256];
  long row = blockIdx.x;
  int t = threadIdx.x;
  float v = x[row * DD + t];
  red[t] = v;
  __syncthreads();
  for (int s = 128; s > 0; s >>= 1) {
    if (t < s) red[t] += red[t + s];
    __syncthreads();
  }
  float mean = red[0] * (1.0f / DD);
  __syncthreads();
  float d = v - mean;
  red[t] = d * d;
  __syncthreads();
  for (int s = 128; s > 0; s >>= 1) {
    if (t < s) red[t] += red[t + s];
    __syncthreads();
  }
  float var = red[0] * (1.0f / DD);
  y[row * DD + t] = d * rsqrtf(var + 1e-5f) * g[t] + be[t];
}

// ---------------- softmax over rows of 128 (attention P) ----------------
__global__ void __launch_bounds__(128) k_softmax128(const float* __restrict__ s,
                                                    h16* __restrict__ p, float alpha) {
  __shared__ float red[128];
  long row = blockIdx.x;
  int t = threadIdx.x;
  float v = s[row * 128 + t] * alpha;
  red[t] = v;
  __syncthreads();
  for (int st = 64; st > 0; st >>= 1) {
    if (t < st) red[t] = fmaxf(red[t], red[t + st]);
    __syncthreads();
  }
  float mx = red[0];
  __syncthreads();
  float e = __expf(v - mx);
  red[t] = e;
  __syncthreads();
  for (int st = 64; st > 0; st >>= 1) {
    if (t < st) red[t] += red[t + st];
    __syncthreads();
  }
  p[row * 128 + t] = (h16)(e / red[0]);
}

// ---------------- converts ----------------
__global__ void k_cvt(const float* __restrict__ x, h16* __restrict__ y, int n) {
  int i = blockIdx.x * 256 + threadIdx.x;
  if (i < n) y[i] = (h16)x[i];
}
__global__ void k_add_cvt(const float* __restrict__ a, const float* __restrict__ b,
                          h16* __restrict__ y, int n) {
  int i = blockIdx.x * 256 + threadIdx.x;
  if (i < n) y[i] = (h16)(a[i] + b[i]);
}

// ---------------- fused relation-attention branch ----------------
__device__ __forceinline__ void store_relu_h(v8f acc, h16* __restrict__ dst,
                                             const float* __restrict__ bias,
                                             int mt, int nt) {
  int l = threadIdx.x & 31;
  int n = nt * 16 + (l & 15);
  int off = (l < 16) ? 0 : 8;
  float bb = bias[n];
#pragma unroll
  for (int j = 0; j < 8; ++j) {
    float v = acc[j] + bb;
    v = v > 0.f ? v : 0.f;
    dst[(mt * 16 + off + j) * DD + n] = (h16)v;
  }
}

__device__ __forceinline__ void accum_soft(v8f sa, v8f va,
                                           const float* __restrict__ b2,
                                           const float* __restrict__ bv,
                                           float* __restrict__ s_den,
                                           float* __restrict__ s_acc, int nt) {
  int l = threadIdx.x & 31;
  int n = nt * 16 + (l & 15);
  float sb = b2[n], vb2 = bv[n];
  float se = 0.f, sev = 0.f;
#pragma unroll
  for (int j = 0; j < 8; ++j) {
    float e = __expf((sa[j] + sb) * 0.0625f);  // / sqrt(256)
    se += e;
    sev += e * (va[j] + vb2);
  }
  se += __shfl_xor(se, 16);
  sev += __shfl_xor(sev, 16);
  if (l < 16) {
    atomicAdd(&s_den[n], se);
    atomicAdd(&s_acc[n], sev);
  }
}

// One workgroup per (q,b). TDM double-buffers 32-row chunks of relative_pos into
// LDS; per chunk: build x/y f16 tiles, h=relu(x@W1^T+b1), sim=h@W2^T+b2,
// v2=y@Wv^T+bv (2x2-blocked WMMA), accumulate sum(e), sum(e*v2) per feature.
__global__ void __launch_bounds__(256) k_rel(
    const float* __restrict__ t2b, const float* __restrict__ mem,
    const float* __restrict__ rel, const h16* __restrict__ W1,
    const float* __restrict__ b1, const h16* __restrict__ W2,
    const float* __restrict__ b2, const h16* __restrict__ Wv,
    const float* __restrict__ bv, const float* __restrict__ o_w,
    const float* __restrict__ o_b, float* __restrict__ out) {
  int q = blockIdx.x, b = blockIdx.y;
  __shared__ __attribute__((aligned(16))) h16 xh[32 * DD];
  __shared__ __attribute__((aligned(16))) h16 yh[32 * DD];
  __shared__ __attribute__((aligned(16))) h16 hh[32 * DD];
#if USE_TDM
  __shared__ __attribute__((aligned(16))) float stage[2 * 32 * DD];  // 64KB
#endif
  __shared__ float s_den[DD], s_acc[DD], s_t[DD];
  int tid = threadIdx.x;
  int l = tid & 31, wid = tid >> 5;
  s_den[tid] = 0.f;
  s_acc[tid] = 0.f;
  const float* t2row = t2b + ((long)q * BB + b) * DD;
  const float* relbase = rel + (((long)q * CC) * BB + b) * DD;

#if USE_TDM
  if (wid == 0)
    tdm_load_2d(relbase, (unsigned)(uintptr_t)(void*)&stage[0], 32, DD, BB * DD);
#endif
  __syncthreads();

  for (int ci = 0; ci < CC / 32; ++ci) {
    int c0 = ci * 32;
    int buf = ci & 1;
#if USE_TDM
    if (wid == 0) {
      if (ci + 1 < CC / 32) {
        tdm_load_2d(relbase + (long)(c0 + 32) * BB * DD,
                    (unsigned)(uintptr_t)(void*)&stage[(buf ^ 1) * 32 * DD], 32, DD,
                    BB * DD);
        __builtin_amdgcn_s_wait_tensorcnt(1);  // current chunk landed
      } else {
        __builtin_amdgcn_s_wait_tensorcnt(0);
      }
    }
#endif
    __syncthreads();
    // build phase: 256 threads cover 32 rows x 256 cols
    {
      int r = tid >> 3;
      int f0 = (tid & 7) * 32;
      int c = c0 + r;
#if USE_TDM
      const float* relp = &stage[buf * 32 * DD + r * DD + f0];
#else
      const float* relp = relbase + ((long)c * BB) * DD + f0;
#endif
      const float* memp = mem + ((long)c * BB + b) * DD + f0;
      const float* tp = t2row + f0;
#pragma unroll
      for (int i = 0; i < 32; i += 4) {
        float4 rv = *(const float4*)(relp + i);
        float4 mv = *(const float4*)(memp + i);
        float4 tv = *(const float4*)(tp + i);
        int base = r * DD + f0 + i;
        xh[base + 0] = (h16)(tv.x - mv.x + rv.x);
        xh[base + 1] = (h16)(tv.y - mv.y + rv.y);
        xh[base + 2] = (h16)(tv.z - mv.z + rv.z);
        xh[base + 3] = (h16)(tv.w - mv.w + rv.w);
        yh[base + 0] = (h16)(mv.x + rv.x);
        yh[base + 1] = (h16)(mv.y + rv.y);
        yh[base + 2] = (h16)(mv.z + rv.z);
        yh[base + 3] = (h16)(mv.w + rv.w);
      }
    }
    __syncthreads();
    // GEMM1 (2x2 blocked): wave w owns mt {0,1} x nt {2w, 2w+1}
    {
      int nt0 = wid * 2, nt1 = nt0 + 1;
      v8f a00 = {}, a01 = {}, a10 = {}, a11 = {};
#pragma unroll
      for (int kt = 0; kt < 8; ++kt) {
        v16h b0 = load_bfrag(W1 + nt0 * 16 * DD, DD, kt * 32);
        v16h b1f = load_bfrag(W1 + nt1 * 16 * DD, DD, kt * 32);
        v16h x0 = load_afrag(xh, DD, kt * 32);
        v16h x1 = load_afrag(xh + 16 * DD, DD, kt * 32);
        a00 = wmma_f16(x0, b0, a00);
        a01 = wmma_f16(x0, b1f, a01);
        a10 = wmma_f16(x1, b0, a10);
        a11 = wmma_f16(x1, b1f, a11);
      }
      store_relu_h(a00, hh, b1, 0, nt0);
      store_relu_h(a01, hh, b1, 0, nt1);
      store_relu_h(a10, hh, b1, 1, nt0);
      store_relu_h(a11, hh, b1, 1, nt1);
    }
    __syncthreads();
    // GEMM2 (sim = h@W2^T) + GEMM3 (v2 = y@Wv^T), 2x2 blocked, then accumulate
    {
      int nt0 = wid * 2, nt1 = nt0 + 1;
      v8f s00 = {}, s01 = {}, s10 = {}, s11 = {};
#pragma unroll
      for (int kt = 0; kt < 8; ++kt) {
        v16h b0 = load_bfrag(W2 + nt0 * 16 * DD, DD, kt * 32);
        v16h b1f = load_bfrag(W2 + nt1 * 16 * DD, DD, kt * 32);
        v16h h0 = load_afrag(hh, DD, kt * 32);
        v16h h1 = load_afrag(hh + 16 * DD, DD, kt * 32);
        s00 = wmma_f16(h0, b0, s00);
        s01 = wmma_f16(h0, b1f, s01);
        s10 = wmma_f16(h1, b0, s10);
        s11 = wmma_f16(h1, b1f, s11);
      }
      v8f v00 = {}, v01 = {}, v10 = {}, v11 = {};
#pragma unroll
      for (int kt = 0; kt < 8; ++kt) {
        v16h b0 = load_bfrag(Wv + nt0 * 16 * DD, DD, kt * 32);
        v16h b1f = load_bfrag(Wv + nt1 * 16 * DD, DD, kt * 32);
        v16h y0 = load_afrag(yh, DD, kt * 32);
        v16h y1 = load_afrag(yh + 16 * DD, DD, kt * 32);
        v00 = wmma_f16(y0, b0, v00);
        v01 = wmma_f16(y0, b1f, v01);
        v10 = wmma_f16(y1, b0, v10);
        v11 = wmma_f16(y1, b1f, v11);
      }
      accum_soft(s00, v00, b2, bv, s_den, s_acc, nt0);
      accum_soft(s01, v01, b2, bv, s_den, s_acc, nt1);
      accum_soft(s10, v10, b2, bv, s_den, s_acc, nt0);
      accum_soft(s11, v11, b2, bv, s_den, s_acc, nt1);
    }
    __syncthreads();
  }
  s_t[tid] = s_acc[tid] / s_den[tid];
  __syncthreads();
  // o-projection epilogue (tiny): out = relu(t @ o_w^T + o_b) + tgt2
  {
    int n = tid;
    const float* wr = o_w + (long)n * DD;
    float v = 0.f;
#pragma unroll 8
    for (int f = 0; f < DD; ++f) v += s_t[f] * wr[f];
    v += o_b[n];
    v = v > 0.f ? v : 0.f;
    out[((long)q * BB + b) * DD + n] = v + t2row[n];
  }
}

extern "C" void kernel_launch(void* const* d_in, const int* in_sizes, int n_in,
                              void* d_out, int out_size, void* d_ws, size_t ws_size,
                              hipStream_t stream) {
  const float* tgt = (const float*)d_in[0];
  const float* mem = (const float*)d_in[1];
  const float* qpos = (const float*)d_in[2];
  const float* rel = (const float*)d_in[3];
  const float* ln1g = (const float*)d_in[4];
  const float* ln1b = (const float*)d_in[5];
  const float* ln2g = (const float*)d_in[6];
  const float* ln2b = (const float*)d_in[7];
  const float* ln3g = (const float*)d_in[8];
  const float* ln3b = (const float*)d_in[9];
  const float* inw = (const float*)d_in[10];
  const float* inb = (const float*)d_in[11];
  const float* aow = (const float*)d_in[12];
  const float* aob = (const float*)d_in[13];
  const float* m1w = (const float*)d_in[14];
  const float* m1b = (const float*)d_in[15];
  const float* m2w = (const float*)d_in[16];
  const float* m2b = (const float*)d_in[17];
  const float* vw = (const float*)d_in[18];
  const float* vb = (const float*)d_in[19];
  const float* ow = (const float*)d_in[20];
  const float* ob = (const float*)d_in[21];
  const float* f1w = (const float*)d_in[22];
  const float* f1b = (const float*)d_in[23];
  const float* f2w = (const float*)d_in[24];
  const float* f2b = (const float*)d_in[25];
  float* out = (float*)d_out;

  char* ws = (char*)d_ws;
  size_t off = 0;
  auto alloc = [&](size_t bytes) {
    void* p = ws + off;
    off += bytes;
    off = (off + 255) & ~(size_t)255;
    return p;
  };
  const int QB = QQ * BB;  // 512 rows
  float* t2a = (float*)alloc(QB * DD * 4);
  float* t2b = (float*)alloc(QB * DD * 4);
  float* tgt1 = (float*)alloc(QB * DD * 4);
  float* tgt3 = (float*)alloc(QB * DD * 4);
  float* t4 = (float*)alloc(QB * DD * 4);
  float* scores = (float*)alloc((size_t)BB * NH * 128 * 128 * 4);
  h16* inw_h = (h16*)alloc(3 * DD * DD * 2);
  h16* aow_h = (h16*)alloc(DD * DD * 2);
  h16* m1_h = (h16*)alloc(DD * DD * 2);
  h16* m2_h = (h16*)alloc(DD * DD * 2);
  h16* vw_h = (h16*)alloc(DD * DD * 2);
  h16* f1_h = (h16*)alloc(FFD * DD * 2);
  h16* f2_h = (h16*)alloc(DD * FFD * 2);
  h16* qk_h = (h16*)alloc(QB * DD * 2);
  h16* t2a_h = (h16*)alloc(QB * DD * 2);
  h16* qkp_h = (h16*)alloc(QB * 2 * DD * 2);
  h16* vp_h = (h16*)alloc(QB * DD * 2);
  h16* P_h = (h16*)alloc((size_t)BB * NH * 128 * 128 * 2);
  h16* oattn_h = (h16*)alloc(QB * DD * 2);
  h16* t4_h = (h16*)alloc(QB * DD * 2);
  h16* ffh = (h16*)alloc((size_t)QB * FFD * 2);

  auto cvt = [&](const float* s, h16* d, int n) {
    k_cvt<<<(n + 255) / 256, 256, 0, stream>>>(s, d, n);
  };
  cvt(inw, inw_h, 3 * DD * DD);
  cvt(aow, aow_h, DD * DD);
  cvt(m1w, m1_h, DD * DD);
  cvt(m2w, m2_h, DD * DD);
  cvt(vw, vw_h, DD * DD);
  cvt(f1w, f1_h, FFD * DD);
  cvt(f2w, f2_h, DD * FFD);

  // ---- self-attention (pre-norm) ----
  k_ln<<<QB, 256, 0, stream>>>(tgt, ln1g, ln1b, t2a);
  cvt(t2a, t2a_h, QB * DD);
  k_add_cvt<<<(QB * DD + 255) / 256, 256, 0, stream>>>(t2a, qpos, qk_h, QB * DD);

  // [qp|kp] = qk @ inW[0:512]^T + b : M=512 N=512 K=256
  k_gemm<<<dim3(32, 4, 1), 256, 0, stream>>>(
      qk_h, 0, 0, 0, DD, inw_h, 0, 0, 0, DD, 1, inb, nullptr, 0, 0, 0, 0, nullptr,
      qkp_h, 0, 0, 0, 2 * DD, QB, 2 * DD, DD, 1, 0, 1.0f);
  // vp = tgt2 @ inW[512:768]^T + b : M=512 N=256 K=256
  k_gemm<<<dim3(16, 4, 1), 256, 0, stream>>>(
      t2a_h, 0, 0, 0, DD, inw_h, (long)2 * DD * DD, 0, 0, DD, 1, inb + 2 * DD,
      nullptr, 0, 0, 0, 0, nullptr, vp_h, 0, 0, 0, DD, QB, DD, DD, 1, 0, 1.0f);
  // scores[b,h,l,s] = qh @ kh^T : per-head, M=N=128, K=32, z=B*H
  k_gemm<<<dim3(8, 1, BB * NH), 256, 0, stream>>>(
      qkp_h, 0, 2 * DD, HDim, BB * 2 * DD, qkp_h, DD, 2 * DD, HDim, BB * 2 * DD, 1,
      nullptr, nullptr, 0, 0, 0, 0, scores, nullptr, 0, (long)NH * 128 * 128,
      128 * 128, 128, 128, 128, HDim, NH, 0, 1.0f);
  k_softmax128<<<BB * NH * 128, 128, 0, stream>>>(scores, P_h, 0.17677669529663687f);
  // o = P @ V : M=128 N=32 K=128, strided B (V is s-major)
  k_gemm<<<dim3(2, 1, BB * NH), 256, 0, stream>>>(
      P_h, 0, (long)NH * 128 * 128, 128 * 128, 128, vp_h, 0, DD, HDim, 1, BB * DD,
      nullptr, nullptr, 0, 0, 0, 0, nullptr, oattn_h, 0, DD, HDim, BB * DD, 128,
      HDim, 128, NH, 0, 1.0f);
  // tgt1 = tgt + o @ aow^T + aob
  k_gemm<<<dim3(16, 4, 1), 256, 0, stream>>>(
      oattn_h, 0, 0, 0, DD, aow_h, 0, 0, 0, DD, 1, aob, tgt, 0, 0, 0, DD, tgt1,
      nullptr, 0, 0, 0, DD, QB, DD, DD, 1, 0, 1.0f);

  // ---- relation-attention branch (the big one) ----
  k_ln<<<QB, 256, 0, stream>>>(tgt1, ln2g, ln2b, t2b);
  k_rel<<<dim3(QQ, BB, 1), 256, 0, stream>>>(t2b, mem, rel, m1_h, m1b, m2_h, m2b,
                                             vw_h, vb, ow, ob, tgt3);

  // ---- FFN ----
  k_ln<<<QB, 256, 0, stream>>>(tgt3, ln3g, ln3b, t4);
  cvt(t4, t4_h, QB * DD);
  k_gemm<<<dim3(64, 4, 1), 256, 0, stream>>>(
      t4_h, 0, 0, 0, DD, f1_h, 0, 0, 0, DD, 1, f1b, nullptr, 0, 0, 0, 0, nullptr,
      ffh, 0, 0, 0, FFD, QB, FFD, DD, 1, 1, 1.0f);
  k_gemm<<<dim3(16, 4, 1), 256, 0, stream>>>(
      ffh, 0, 0, 0, FFD, f2_h, 0, 0, 0, FFD, 1, f2b, tgt3, 0, 0, 0, DD, out,
      nullptr, 0, 0, 0, DD, QB, DD, FFD, 1, 0, 1.0f);
}
